// DimeNet_27994596835752
// MI455X (gfx1250) — matrix operand order repaired and checked by reading, not compiled, not run on hardware
//
#include <hip/hip_runtime.h>
#include <math.h>

// ---------------- problem constants ----------------
#define E_N   200000
#define A_N   1000000
#define H_N   128
#define BD_N  8
#define NR_N  6
#define NS_N  7
#define MIN_N 128

#define AS1 __attribute__((address_space(1)))

typedef __bf16 bf16_t;
typedef __attribute__((ext_vector_type(16))) __bf16       v16bf;
typedef __attribute__((ext_vector_type(8)))  __bf16       v8bf;
typedef __attribute__((ext_vector_type(8)))  float        v8f;
typedef __attribute__((ext_vector_type(4)))  unsigned int u32x4;
typedef __attribute__((ext_vector_type(4)))  float        f32x4;

// A/B fragment: 16 bf16 per lane (32 bytes)
union ABfrag { v16bf v; u32x4 q[2]; };
union Q8     { v8bf  v; u32x4 q;    };
static_assert(sizeof(v16bf) == 32, "v16bf must be 32B");
static_assert(sizeof(v8bf)  == 16, "v8bf must be 16B");

// Opaque pointer laundering into the GLOBAL address space:
//  - prevents LICM from hoisting weight-fragment loads out of the tile loop
//    (which caused scratch spills/reloads)
//  - bounds in-flight B fragments to one K-step (8 frags = 64 VGPRs)
//  - addrspace(1) result => global_load_b128 (LOADcnt only), not flat_load
//    (which also ticks DScnt and shares the LDS path)
__device__ __forceinline__ AS1 const bf16_t* launder_g(const bf16_t* p) {
  unsigned long long x = (unsigned long long)p;
  asm volatile("" : "+v"(x));
  return (AS1 const bf16_t*)x;
}

// ---------------------------------------------------------------------------
// Weight packing: store each K x N (K mult of 32, N mult of 16) matrix as
// WMMA-B fragments. Fragment (kb, nb) covers rows k = 32*kb.., cols n=16*nb..
// B layout used here (16-bit, 32x16, contiguous k; A matches): lanes 0-15 hold K=0..15 (2 per VGPR),
// lanes 16-31 hold K=16..31; lane%16 selects column n.
// Element j (0..15) of lane l corresponds to k_local = (l<16?0:16)+j, n = l%16.
// ---------------------------------------------------------------------------
__global__ void pack_weights_kernel(const float* __restrict__ Wbil,
                                    const float* __restrict__ Wsrc,
                                    const float* __restrict__ Wtgt,
                                    const float* __restrict__ RbW,
                                    const float* __restrict__ Wskip,
                                    const float* __restrict__ RaW,
                                    bf16_t* __restrict__ wbig_p,
                                    bf16_t* __restrict__ wsrc_p,
                                    bf16_t* __restrict__ wmlp_p) {
  int tid  = blockIdx.x * blockDim.x + threadIdx.x;
  int frag = tid >> 5, lane = tid & 31;
  if (frag >= 544) return;
  int klo0 = (lane < 16) ? 0 : 16;
  int n_l  = lane & 15;

  if (frag < 256) {
    // W_bil reshaped: Wbig[k = b*128 + h][n = i] = W_bil[i][b][h]
    int kb = frag >> 3, nb = frag & 7;
    bf16_t* dst = wbig_p + (size_t)frag * 512 + lane * 16;
    #pragma unroll
    for (int j = 0; j < 16; ++j) {
      int k = kb * 32 + klo0 + j;
      int b = k >> 7, h = k & 127;
      int n = nb * 16 + n_l;
      dst[j] = (bf16_t)Wbil[((size_t)n * BD_N + b) * H_N + h];
    }
  } else if (frag < 288) {
    int f = frag - 256, kb = f >> 3, nb = f & 7;
    bf16_t* dst = wsrc_p + (size_t)f * 512 + lane * 16;
    #pragma unroll
    for (int j = 0; j < 16; ++j) {
      int k = kb * 32 + klo0 + j, n = nb * 16 + n_l;
      dst[j] = (bf16_t)Wsrc[(size_t)k * H_N + n];
    }
  } else {
    int f = frag - 288, mat = f >> 5, ff = f & 31;
    int kb = ff >> 3, nb = ff & 7;
    const float* src;
    switch (mat) {
      case 0: src = Wtgt;                      break; // W_tgt
      case 1: src = RbW;                       break; // res_before L1
      case 2: src = RbW + H_N * H_N;           break; // res_before L2
      case 3: src = Wskip;                     break; // W_skip
      case 4: src = RaW;                       break; // res_after0 L1
      case 5: src = RaW + 1 * MIN_N * MIN_N;   break; // res_after0 L2
      case 6: src = RaW + 2 * MIN_N * MIN_N;   break; // res_after1 L1
      default: src = RaW + 3 * MIN_N * MIN_N;  break; // res_after1 L2
    }
    bf16_t* dst = wmlp_p + (size_t)f * 512 + lane * 16;
    #pragma unroll
    for (int j = 0; j < 16; ++j) {
      int k = kb * 32 + klo0 + j, n = nb * 16 + n_l;
      dst[j] = (bf16_t)src[(size_t)k * 128 + n];
    }
  }
}

// ---------------------------------------------------------------------------
// d = dist_rep @ W_dist  (E x 6 @ 6 x 128), stored bf16 (L2-resident table)
// ---------------------------------------------------------------------------
__global__ void prep_d_kernel(const float* __restrict__ dist,
                              const float* __restrict__ Wdist,
                              bf16_t* __restrict__ d_bf) {
  int tid = blockIdx.x * blockDim.x + threadIdx.x;
  int e = tid >> 5, lane = tid & 31;
  if (e >= E_N) return;
  float dr[NR_N];
  #pragma unroll
  for (int r = 0; r < NR_N; ++r) dr[r] = dist[(size_t)e * NR_N + r];
  int h0 = lane * 4;
  #pragma unroll
  for (int c = 0; c < 4; ++c) {
    float acc = 0.f;
    #pragma unroll
    for (int r = 0; r < NR_N; ++r) acc += dr[r] * Wdist[r * H_N + h0 + c];
    d_bf[(size_t)e * H_N + h0 + c] = (bf16_t)acc;
  }
}

// ---------------------------------------------------------------------------
// Wave-level helpers (each wave owns one 16-row x 128-col tile in the 8x v8f
// accumulator layout: element (r,lane) = row r + 8*(lane/16), col nb*16+lane%16)
// ---------------------------------------------------------------------------
__device__ __forceinline__ void wv_zero(v8f acc[8]) {
  #pragma unroll
  for (int nb = 0; nb < 8; ++nb)
    #pragma unroll
    for (int r = 0; r < 8; ++r) acc[nb][r] = 0.f;
}

__device__ __forceinline__ void wv_gemm(const bf16_t* __restrict__ X,   // [16][128] LDS bf16
                                        const bf16_t* __restrict__ wp,  // 32 packed fragments
                                        v8f acc[8], int lane) {
  int row   = lane & 15;
  int kbase = (lane < 16) ? 0 : 16;  // A layout: lanes 0-15 K=0..15, lanes 16-31 K=16..31
  #pragma unroll
  for (int kb = 0; kb < 4; ++kb) {
    // fresh opaque global pointer per K-step: max 8 B-frags in flight
    AS1 const bf16_t* wkb = launder_g(wp + (size_t)kb * 8 * 512);
    ABfrag a;
    const bf16_t* rp = X + row * 128 + kb * 32 + kbase;
    a.q[0] = *(const u32x4*)(rp);
    a.q[1] = *(const u32x4*)(rp + 8);
    #pragma unroll
    for (int nb = 0; nb < 8; ++nb) {
      v16bf bm = *(AS1 const v16bf*)(wkb + (size_t)nb * 512 + lane * 16);
      acc[nb] = __builtin_amdgcn_wmma_f32_16x16x32_bf16(
          false, a.v, false, bm, (short)0, acc[nb], false, false);
    }
  }
}

__device__ __forceinline__ void wv_bias(v8f acc[8], const float* __restrict__ b, int lane) {
  int col = lane & 15;
  #pragma unroll
  for (int nb = 0; nb < 8; ++nb) {
    float bv = b[nb * 16 + col];
    #pragma unroll
    for (int r = 0; r < 8; ++r) acc[nb][r] += bv;
  }
}

__device__ __forceinline__ void wv_silu(v8f acc[8]) {
  #pragma unroll
  for (int nb = 0; nb < 8; ++nb)
    #pragma unroll
    for (int r = 0; r < 8; ++r) {
      float v = acc[nb][r];
      acc[nb][r] = v / (1.f + __expf(-v));
    }
}

__device__ __forceinline__ void wv_add(v8f x[8], const v8f h[8]) {
  #pragma unroll
  for (int nb = 0; nb < 8; ++nb)
    #pragma unroll
    for (int r = 0; r < 8; ++r) x[nb][r] += h[nb][r];
}

// write accumulator tile back to LDS as bf16 rows (next GEMM's A operand).
// Within-wave DS ops are in-order, so no barrier needed.
__device__ __forceinline__ void wv_restage(bf16_t* __restrict__ X, const v8f acc[8], int lane) {
  int col = lane & 15;
  int rowoff = (lane < 16) ? 0 : 8;
  #pragma unroll
  for (int nb = 0; nb < 8; ++nb)
    #pragma unroll
    for (int r = 0; r < 8; ++r)
      X[(rowoff + r) * 128 + nb * 16 + col] = (bf16_t)acc[nb][r];
}

// ---------------------------------------------------------------------------
// msg_src = bf16(message @ W_src + b_src)   [E,128] — WMMA GEMM
// ---------------------------------------------------------------------------
__global__ void msgsrc_kernel(const float* __restrict__ msg,
                              const bf16_t* __restrict__ wsrc_p,
                              const float* __restrict__ bsrc,
                              bf16_t* __restrict__ out) {
  __shared__ __align__(16) bf16_t xs[8][16 * 128];
  int wave = threadIdx.x >> 5, lane = threadIdx.x & 31;
  bf16_t* X = xs[wave];
  int ntiles = E_N / 16;
  for (int tile = blockIdx.x * 8 + wave; tile < ntiles; tile += gridDim.x * 8) {
    int e0 = tile * 16;
    for (int t = lane; t < 512; t += 32) {            // stage 16x128 f32 -> bf16
      int row = t >> 5, c4 = (t & 31) * 4;
      f32x4 v = *(AS1 const f32x4*)(msg + (size_t)(e0 + row) * MIN_N + c4);
      bf16_t* q = X + row * 128 + c4;
      q[0] = (bf16_t)v.x; q[1] = (bf16_t)v.y; q[2] = (bf16_t)v.z; q[3] = (bf16_t)v.w;
    }
    v8f acc[8];
    wv_zero(acc);
    wv_gemm(X, wsrc_p, acc, lane);
    wv_bias(acc, bsrc, lane);
    int col = lane & 15, rowoff = (lane < 16) ? 0 : 8;
    #pragma unroll
    for (int nb = 0; nb < 8; ++nb)
      #pragma unroll
      for (int r = 0; r < 8; ++r)
        out[(size_t)(e0 + rowoff + r) * H_N + nb * 16 + col] = (bf16_t)acc[nb][r];
  }
}

// ---------------------------------------------------------------------------
// Angle kernel: per 16-angle tile (one wave):
//   a[m,b]   = angle_rep[m].flat(42) @ W_angle            (VALU, lanes 0-15)
//   s[m,h]   = msg_src[src_m,h] * d[tgt_m,h]              (v_pk_mul_bf16, LDS)
//   out[m,i] = sum_{k=b*128+h} a[m,b]*s[m,h] * Wbig[k,i]  (32 K-steps of WMMA;
//              b is constant inside each 32-wide K block -> pk-scale A-frag)
//   agg[tgt_m, i] += out[m,i]                             (global f32 atomics)
// ---------------------------------------------------------------------------
__global__ void angle_kernel(const int*   __restrict__ aidx,
                             const float* __restrict__ arep,
                             const float* __restrict__ Wangle_g,
                             const bf16_t* __restrict__ msrc,
                             const bf16_t* __restrict__ dbf,
                             const bf16_t* __restrict__ wbig_p,
                             float* __restrict__ agg) {
  __shared__ __align__(16) bf16_t s[8][16 * 128];
  __shared__ float asc[8][16][8];
  __shared__ int   sg[8][16];
  __shared__ int   tg[8][16];
  __shared__ float Wang[NS_N * NR_N * BD_N];  // 42*8
  for (int i = threadIdx.x; i < NS_N * NR_N * BD_N; i += blockDim.x)
    Wang[i] = Wangle_g[i];
  __syncthreads();

  int wave = threadIdx.x >> 5, lane = threadIdx.x & 31;
  int ntiles = A_N / 16;
  for (int tile = blockIdx.x * 8 + wave; tile < ntiles; tile += gridDim.x * 8) {
    int a0 = tile * 16;
    if (lane < 16) {
      sg[wave][lane] = aidx[a0 + lane];
      tg[wave][lane] = aidx[A_N + a0 + lane];
      // a-vector: 42x8 matvec per angle
      float av[8];
      #pragma unroll
      for (int b = 0; b < 8; ++b) av[b] = 0.f;
      const float* ar = arep + (size_t)(a0 + lane) * (NS_N * NR_N);
      for (int j = 0; j < NS_N * NR_N; ++j) {
        float x = ar[j];
        #pragma unroll
        for (int b = 0; b < 8; ++b) av[b] += x * Wang[j * 8 + b];
      }
      #pragma unroll
      for (int b = 0; b < 8; ++b) asc[wave][lane][b] = av[b];
    }
    // stage s rows: 16B chunks, packed bf16 multiply (gathers hit L2 tables)
    for (int t = lane; t < 256; t += 32) {
      int row = t >> 4, c8 = (t & 15) * 8;
      int se = sg[wave][row], te = tg[wave][row];
      Q8 m, d2, p;
      m.q  = *(AS1 const u32x4*)(msrc + (size_t)se * H_N + c8);
      d2.q = *(AS1 const u32x4*)(dbf  + (size_t)te * H_N + c8);
      p.v = m.v * d2.v;                          // 4x v_pk_mul_bf16
      *(u32x4*)(&s[wave][row * 128 + c8]) = p.q; // ds_store_b128
    }

    v8f acc[8];
    wv_zero(acc);
    int row = lane & 15;
    int kbase = (lane < 16) ? 0 : 16;
    for (int kb = 0; kb < 32; ++kb) {
      int b = kb >> 2;                       // bilinear index for this K block
      bf16_t sc = (bf16_t)asc[wave][row][b];
      // fresh opaque global pointer per K-step (no hoist/spill of Wbig stream)
      AS1 const bf16_t* wkb = launder_g(wbig_p + (size_t)kb * 8 * 512);
      ABfrag a, asl;
      const bf16_t* rp = &s[wave][row * 128 + (kb & 3) * 32 + kbase];
      a.q[0] = *(const u32x4*)(rp);
      a.q[1] = *(const u32x4*)(rp + 8);
      asl.v = a.v * sc;                      // 8x v_pk_mul_bf16 (splat scale)
      #pragma unroll
      for (int nb = 0; nb < 8; ++nb) {
        v16bf bm = *(AS1 const v16bf*)(wkb + (size_t)nb * 512 + lane * 16);
        acc[nb] = __builtin_amdgcn_wmma_f32_16x16x32_bf16(
            false, asl.v, false, bm, (short)0, acc[nb], false, false);
      }
    }
    // scatter-add into agg via global f32 atomics (L2-resident, relaxed)
    int rowoff = (lane < 16) ? 0 : 8;
    int col = lane & 15;
    #pragma unroll
    for (int r = 0; r < 8; ++r) {
      int te = tg[wave][rowoff + r];
      AS1 float* base = (AS1 float*)(agg + (size_t)te * H_N);
      #pragma unroll
      for (int nb = 0; nb < 8; ++nb)
        __hip_atomic_fetch_add(base + nb * 16 + col, acc[nb][r],
                               __ATOMIC_RELAXED, __HIP_MEMORY_SCOPE_AGENT);
    }
  }
}

// ---------------------------------------------------------------------------
// Fused edge MLP: x = agg + msg@W_tgt + b; res_before; skip(+msg); 2x res_after
// One wave per 16-edge tile; activation tile stays in accumulator registers,
// restaged to LDS bf16 between the 8 chained WMMA GEMMs.
// ---------------------------------------------------------------------------
__global__ void mlp_kernel(const float* __restrict__ msg,
                           const float* __restrict__ agg,
                           const bf16_t* __restrict__ wmlp_p,
                           const float* __restrict__ btgt,
                           const float* __restrict__ rbb,
                           const float* __restrict__ bskip,
                           const float* __restrict__ rab,
                           float* __restrict__ out) {
  __shared__ __align__(16) bf16_t xs[8][16 * 128];
  int wave = threadIdx.x >> 5, lane = threadIdx.x & 31;
  bf16_t* X = xs[wave];
  int col = lane & 15, rowoff = (lane < 16) ? 0 : 8;
  int ntiles = E_N / 16;
  for (int tile = blockIdx.x * 8 + wave; tile < ntiles; tile += gridDim.x * 8) {
    int e0 = tile * 16;
    for (int t = lane; t < 512; t += 32) {          // stage message tile bf16
      int row = t >> 5, c4 = (t & 31) * 4;
      f32x4 v = *(AS1 const f32x4*)(msg + (size_t)(e0 + row) * MIN_N + c4);
      bf16_t* q = X + row * 128 + c4;
      q[0] = (bf16_t)v.x; q[1] = (bf16_t)v.y; q[2] = (bf16_t)v.z; q[3] = (bf16_t)v.w;
    }
    v8f x[8], h[8];
    // x = agg + msg @ W_tgt + b_tgt
    wv_zero(x); wv_gemm(X, wmlp_p + 0 * 16384, x, lane); wv_bias(x, btgt, lane);
    #pragma unroll
    for (int nb = 0; nb < 8; ++nb)
      #pragma unroll
      for (int r = 0; r < 8; ++r)
        x[nb][r] += agg[(size_t)(e0 + rowoff + r) * H_N + nb * 16 + col];
    // residual-before
    wv_restage(X, x, lane);
    wv_zero(h); wv_gemm(X, wmlp_p + 1 * 16384, h, lane); wv_bias(h, rbb + 0,   lane); wv_silu(h);
    wv_restage(X, h, lane);
    wv_zero(h); wv_gemm(X, wmlp_p + 2 * 16384, h, lane); wv_bias(h, rbb + 128, lane); wv_silu(h);
    wv_add(x, h);
    // skip: x = silu(x @ W_skip + b_skip) + message
    wv_restage(X, x, lane);
    wv_zero(h); wv_gemm(X, wmlp_p + 3 * 16384, h, lane); wv_bias(h, bskip, lane); wv_silu(h);
    #pragma unroll
    for (int nb = 0; nb < 8; ++nb)
      #pragma unroll
      for (int r = 0; r < 8; ++r)
        x[nb][r] = h[nb][r] + msg[(size_t)(e0 + rowoff + r) * MIN_N + nb * 16 + col];
    // residual-after 0
    wv_restage(X, x, lane);
    wv_zero(h); wv_gemm(X, wmlp_p + 4 * 16384, h, lane); wv_bias(h, rab + 0,   lane); wv_silu(h);
    wv_restage(X, h, lane);
    wv_zero(h); wv_gemm(X, wmlp_p + 5 * 16384, h, lane); wv_bias(h, rab + 128, lane); wv_silu(h);
    wv_add(x, h);
    // residual-after 1
    wv_restage(X, x, lane);
    wv_zero(h); wv_gemm(X, wmlp_p + 6 * 16384, h, lane); wv_bias(h, rab + 256, lane); wv_silu(h);
    wv_restage(X, h, lane);
    wv_zero(h); wv_gemm(X, wmlp_p + 7 * 16384, h, lane); wv_bias(h, rab + 384, lane); wv_silu(h);
    wv_add(x, h);
    // store f32 output
    #pragma unroll
    for (int nb = 0; nb < 8; ++nb)
      #pragma unroll
      for (int r = 0; r < 8; ++r)
        out[(size_t)(e0 + rowoff + r) * MIN_N + nb * 16 + col] = x[nb][r];
  }
}

// ---------------------------------------------------------------------------
extern "C" void kernel_launch(void* const* d_in, const int* in_sizes, int n_in,
                              void* d_out, int out_size, void* d_ws, size_t ws_size,
                              hipStream_t stream) {
  const float* dist  = (const float*)d_in[0];
  const float* arep  = (const float*)d_in[1];
  const float* msg   = (const float*)d_in[2];
  const int*   aidx  = (const int*)  d_in[3];
  const float* Wdist = (const float*)d_in[4];
  const float* Wang  = (const float*)d_in[5];
  const float* Wsrc  = (const float*)d_in[6];
  const float* bsrc  = (const float*)d_in[7];
  const float* Wtgt  = (const float*)d_in[8];
  const float* btgt  = (const float*)d_in[9];
  const float* Wbil  = (const float*)d_in[10];
  const float* rbW   = (const float*)d_in[11];
  const float* rbb   = (const float*)d_in[12];
  const float* Wskip = (const float*)d_in[13];
  const float* bskip = (const float*)d_in[14];
  const float* raW   = (const float*)d_in[15];
  const float* rab   = (const float*)d_in[16];

  // workspace layout (all offsets 256B aligned)
  char* ws = (char*)d_ws;
  bf16_t* msrc_bf = (bf16_t*)(ws + 0);           // E*128 bf16 = 51.2 MB
  bf16_t* d_bf    = (bf16_t*)(ws + 51200000);    // E*128 bf16 = 51.2 MB
  float*  agg     = (float*) (ws + 102400000);   // E*128 f32  = 102.4 MB
  bf16_t* wbig_p  = (bf16_t*)(ws + 204800000);   // 1024*128 bf16 packed
  bf16_t* wsrc_p  = (bf16_t*)(ws + 205062144);   // 128*128 bf16 packed
  bf16_t* wmlp_p  = (bf16_t*)(ws + 205094912);   // 8 * 128*128 bf16 packed

  (void)hipMemsetAsync(agg, 0, (size_t)E_N * H_N * sizeof(float), stream);

  pack_weights_kernel<<<68, 256, 0, stream>>>(Wbil, Wsrc, Wtgt, rbW, Wskip, raW,
                                              wbig_p, wsrc_p, wmlp_p);
  prep_d_kernel<<<(E_N * 32 + 255) / 256, 256, 0, stream>>>(dist, Wdist, d_bf);
  msgsrc_kernel<<<1563, 256, 0, stream>>>(msg, wsrc_p, bsrc, msrc_bf);
  angle_kernel<<<7813, 256, 0, stream>>>(aidx, arep, Wang, msrc_bf, d_bf, wbig_p, agg);
  mlp_kernel<<<1563, 256, 0, stream>>>(msg, agg, wmlp_p, btgt, rbb, bskip, rab,
                                       (float*)d_out);
}
